// WeightOnlyInt4Linear_25744033972725
// MI455X (gfx1250) — matrix operand order repaired and checked by reading, not compiled, not run on hardware
//
#include <hip/hip_runtime.h>

typedef __attribute__((ext_vector_type(16))) _Float16 v16h;
typedef __attribute__((ext_vector_type(8)))  float    v8f;
typedef __attribute__((ext_vector_type(4)))  float    v4f;
typedef __attribute__((ext_vector_type(4)))  int      v4i;

#define IN_F      4096
#define OUT_F     12288
#define BATCH     8
#define GROUPSIZE 128
#define N_GROUPS  32
#define KSPLIT    4      // waves cooperating on one n-tile along K
#define TILES_PER_BLOCK 2

// One wave computes a 16(n) x 16(b) WMMA tile over a 1024-wide K slice.
// A = dequantized weights (16 rows x 32 K, f16), B = x^T (32 K x 16 batch, f16).
// 4 waves per tile reduce partial accumulators through LDS (deterministic).
__global__ __launch_bounds__(256) void WeightOnlyInt4Linear_wmma_kernel(
    const float* __restrict__ x,
    const int*   __restrict__ qweight,
    const float* __restrict__ scales,
    const float* __restrict__ zeros,
    float* __restrict__ out)
{
    __shared__ float red[8][32 * 8];   // [wave][lane*8 + v], 8 KB

    const int lane = threadIdx.x & 31;
    const int wave = threadIdx.x >> 5;          // 0..7
    const int tile = wave >> 2;                 // 0..1 : n-tile within block
    const int ksp  = wave & (KSPLIT - 1);       // 0..3 : K split index
    const int nbase = (blockIdx.x * TILES_PER_BLOCK + tile) * 16;

    const int half = lane >> 4;                 // lane group (K-half selector)
    const int mrow = lane & 15;                 // A row within tile / B column
    const int n    = nbase + mrow;              // weight row handled by this lane (A)
    const int bb   = mrow;                      // batch row for B (valid if < BATCH)

    const long qrow = (long)n * IN_F;

    v8f c = {};

    const int g0 = ksp * (N_GROUPS / KSPLIT);
    const int g1 = g0 + (N_GROUPS / KSPLIT);

    for (int g = g0; g < g1; ++g) {
        const float s = scales[g * OUT_F + n];
        const float z = zeros [g * OUT_F + n] - 8.0f * s;   // fold (q-8)*s+z -> q*s+z'

        if (g + 1 < g1) {
            // hint: pull next group's weight line toward the caches
            __builtin_prefetch(qweight + qrow + (long)(g + 1) * GROUPSIZE + half * 8, 0, 1);
        }

#pragma unroll
        for (int kk = 0; kk < 4; ++kk) {
            const int k0 = g * GROUPSIZE + kk * 32;

            // ---- A tile: 16x32 dequantized weights (f16) ----
            // half 0 lanes hold K = k0+[0..7]  and k0+[16..23]
            // half 1 lanes hold K = k0+[8..15] and k0+[24..31]
            const v4i* qp = (const v4i*)(qweight + qrow + k0 + half * 8);
            v4i qa0 = __builtin_nontemporal_load(qp + 0);
            v4i qa1 = __builtin_nontemporal_load(qp + 1);
            v4i qb0 = __builtin_nontemporal_load(qp + 4);   // +16 ints
            v4i qb1 = __builtin_nontemporal_load(qp + 5);
            v16h a;
#pragma unroll
            for (int j = 0; j < 4; ++j) {
                a[j]      = (_Float16)fmaf((float)qa0[j], s, z);
                a[4 + j]  = (_Float16)fmaf((float)qa1[j], s, z);
                a[8 + j]  = (_Float16)fmaf((float)qb0[j], s, z);
                a[12 + j] = (_Float16)fmaf((float)qb1[j], s, z);
            }

            // ---- B tile: 32x16 of x^T (f16), zero-padded for b >= BATCH ----
            // half 0 lanes hold K = k0+[0..7] and k0+[16..23], half 1 lanes K = k0+[8..15] and k0+[24..31]
            v16h b = {};
            if (bb < BATCH) {
                const v4f* xp = (const v4f*)(x + (long)bb * IN_F + k0 + half * 8);
                v4f x0 = xp[0], x1 = xp[1], x2 = xp[4], x3 = xp[5];
#pragma unroll
                for (int j = 0; j < 4; ++j) {
                    b[j]      = (_Float16)x0[j];
                    b[4 + j]  = (_Float16)x1[j];
                    b[8 + j]  = (_Float16)x2[j];
                    b[12 + j] = (_Float16)x3[j];
                }
            }

            // D(16n x 16b) += A(16n x 32k) * B(32k x 16b)
            c = __builtin_amdgcn_wmma_f32_16x16x32_f16(
                    false, a, false, b, (short)0, c, false, false);
        }
    }

    // ---- cross-wave K reduction via LDS (deterministic order) ----
    {
        v4f lo = {c[0], c[1], c[2], c[3]};
        v4f hi = {c[4], c[5], c[6], c[7]};
        v4f* dst = (v4f*)&red[wave][lane * 8];
        dst[0] = lo;
        dst[1] = hi;
    }
    __syncthreads();

    if (ksp == 0 && bb < BATCH) {
        v4f lo = {}, hi = {};
#pragma unroll
        for (int t = 0; t < KSPLIT; ++t) {
            const v4f* src = (const v4f*)&red[tile * KSPLIT + t][lane * 8];
            v4f a0 = src[0];
            v4f a1 = src[1];
            lo += a0;
            hi += a1;
        }
        // D layout: lanes 0-15 -> n = nbase+0..7, lanes 16-31 -> n = nbase+8..15,
        // 8 consecutive n per lane across the accumulator VGPRs.
        float* op = out + (long)bb * OUT_F + nbase + half * 8;
        *(v4f*)(op + 0) = lo;
        *(v4f*)(op + 4) = hi;
    }
}

extern "C" void kernel_launch(void* const* d_in, const int* in_sizes, int n_in,
                              void* d_out, int out_size, void* d_ws, size_t ws_size,
                              hipStream_t stream) {
    (void)in_sizes; (void)n_in; (void)out_size; (void)d_ws; (void)ws_size;
    const float* x  = (const float*)d_in[0];
    const int*   qw = (const int*)  d_in[1];
    const float* sc = (const float*)d_in[2];
    const float* zr = (const float*)d_in[3];
    float* out = (float*)d_out;

    dim3 block(256);                                 // 8 wave32s
    dim3 grid(OUT_F / (16 * TILES_PER_BLOCK));       // 384 blocks -> 3072 waves
    hipLaunchKernelGGL(WeightOnlyInt4Linear_wmma_kernel, grid, block, 0, stream,
                       x, qw, sc, zr, out);
}
